// BaselineTransformerLayer_89000312308365
// MI455X (gfx1250) — matrix operand order repaired and checked by reading, not compiled, not run on hardware
//
#include <hip/hip_runtime.h>
#include <cstddef>
#include <cstdint>

// ---------------------------------------------------------------------------
// Types for CDNA5 WMMA (gfx1250, wave32)
// ---------------------------------------------------------------------------
typedef __attribute__((ext_vector_type(16))) __bf16 v16bf;
typedef __attribute__((ext_vector_type(8)))  __bf16 v8bf;
typedef __attribute__((ext_vector_type(8)))  float  v8f;

__device__ inline v16bf cat8(v8bf lo, v8bf hi) {
  return __builtin_shufflevector(lo, hi, 0,1,2,3,4,5,6,7,8,9,10,11,12,13,14,15);
}

__device__ inline v8f wmma_bf16(v16bf a, v16bf b, v8f c) {
  // D = A(16x32) * B(32x16) + C(16x16), f32 accumulate
  return __builtin_amdgcn_wmma_f32_16x16x32_bf16(
      /*neg_a=*/false, a, /*neg_b=*/false, b,
      /*c_mod=*/(short)0, c, /*reuse_a=*/false, /*reuse_b=*/false);
}

// A-matrix fragment (16x32 bf16, MxK). Lane m = lane&15 holds row m:
//   lanes 0-15 : K = [0,8) in VGPR0-3, K = [16,24) in VGPR4-7
//   lanes 16-31: K = [8,16),          K = [24,32)
__device__ inline v16bf load_a_frag(const __bf16* __restrict__ A, int lda,
                                    int m0, int k0, int lane) {
  int row = m0 + (lane & 15);
  int kb  = k0 + ((lane >> 4) << 3);
  const __bf16* p = A + (size_t)row * lda + kb;
  v8bf lo = *(const v8bf*)(p);
  v8bf hi = *(const v8bf*)(p + 16);
  return cat8(lo, hi);
}

__device__ inline float gelu_tanh(float x) {
  float x3 = x * x * x;
  return 0.5f * x * (1.0f + tanhf(0.7978845608028654f * (x + 0.044715f * x3)));
}

// CDNA5 async global->LDS copy (ASYNCcnt path) + wait
__device__ inline void async_copy_b128(uint32_t lds_off, const __bf16* g) {
  asm volatile("global_load_async_to_lds_b128 %0, %1, off"
               :: "v"(lds_off), "v"(g) : "memory");
}
__device__ inline void wait_asynccnt0() {
  asm volatile("s_wait_asynccnt 0x0" ::: "memory");
}

// ---------------------------------------------------------------------------
// Problem constants
// ---------------------------------------------------------------------------
#define SEQ   2048
#define HID   1024
#define FFD   2048
#define NEXP  8
#define NHEAD 16
#define HDIM  64
#define QKVO  1536   // NKV*(q_per_kv+2)*HD = 4*6*64
#define CAPE  640    // ceil(2048*2/8 * 1.25)
#define TSLOT (NEXP*CAPE)  // 5120

// ---------------------------------------------------------------------------
// 4-wave 64(M) x 64(N) WMMA GEMM block:  C = A[M,K] * B[N,K]^T
// B panel staged in double-buffered LDS (8 KB per buffer, K-step 64) via async
// global->LDS copies: 8 WMMAs per wait/barrier, copy of step k+1 overlapped
// with compute of step k, 4x B reuse across the block's waves.
// MODE 0: f32 out   1: f32 out + residual   2: bf16 out   3: gelu -> bf16 out
// rowsPerExpert/bExpertStride: per-expert B selection for batched MoE GEMMs.
// ---------------------------------------------------------------------------
template <int MODE>
__global__ __launch_bounds__(128)
void gemm64x64(const __bf16* __restrict__ A, const __bf16* __restrict__ B,
               int lda, int ldb, void* __restrict__ Cout, int ldc, int Kdim,
               const float* __restrict__ resid, int rowsPerExpert,
               size_t bExpertStride) {
  __shared__ __bf16 bsh[2][64 * 64];   // [buf][col][k], 128B per col row

  int tid = threadIdx.x;
  int lane = tid & 31;
  int w = tid >> 5;
  int mblk = blockIdx.x * 64;
  int m0 = mblk + w * 16;
  int n0 = blockIdx.y * 64;
  const __bf16* Bp = B + (size_t)(mblk / rowsPerExpert) * bExpertStride;

  // Stage one 64x64 bf16 B tile: 512 x 16B chunks, 4 per thread.
  auto issue = [&](int buf, int k0) {
#pragma unroll
    for (int c = 0; c < 4; c++) {
      int chunk = tid * 4 + c;     // 0..511
      int col = chunk >> 3;        // 0..63
      int part = chunk & 7;        // 16B sub-chunk (8 bf16)
      const __bf16* g = Bp + (size_t)(n0 + col) * ldb + k0 + part * 8;
      uint32_t lo = (uint32_t)(uintptr_t)&bsh[buf][col * 64 + part * 8];
      async_copy_b128(lo, g);
    }
  };

  v8f acc[4] = {};
  int buf = 0;
  issue(0, 0);
  int hh = lane >> 4, ln = lane & 15;
  for (int k0 = 0; k0 < Kdim; k0 += 64) {
    wait_asynccnt0();
    __syncthreads();                       // whole tile resident for all waves
    if (k0 + 64 < Kdim) {
      issue(buf ^ 1, k0 + 64);
      // keep the direct-from-global A stream ahead (global_prefetch_b8)
      __builtin_prefetch(A + (size_t)(m0 + ln) * lda + k0 + 64, 0, 1);
    }
#pragma unroll
    for (int kk = 0; kk < 64; kk += 32) {
      v16bf a = load_a_frag(A, lda, m0, k0 + kk, lane);
#pragma unroll
      for (int t = 0; t < 4; t++) {
        const __bf16* bp = &bsh[buf][(t * 16 + ln) * 64 + kk + (hh << 3)];
        v16bf b = cat8(*(const v8bf*)bp, *(const v8bf*)(bp + 16));
        acc[t] = wmma_bf16(a, b, acc[t]);
      }
    }
    buf ^= 1;
  }

#pragma unroll
  for (int t = 0; t < 4; t++) {
#pragma unroll
    for (int j = 0; j < 8; j++) {
      float v = acc[t][j];
      size_t idx = (size_t)(m0 + j + 8 * hh) * ldc + n0 + t * 16 + ln;
      if (MODE == 0) {
        ((float*)Cout)[idx] = v;
      } else if (MODE == 1) {
        ((float*)Cout)[idx] = v + resid[idx];
      } else if (MODE == 2) {
        ((__bf16*)Cout)[idx] = (__bf16)v;
      } else {
        ((__bf16*)Cout)[idx] = (__bf16)gelu_tanh(v);
      }
    }
  }
}

// ---------------------------------------------------------------------------
// Flash-style causal GQA attention. 1 wave per (16-query tile, head).
// qkv layout per row s (QKVO=1536): group g: [4 q heads x64 | k x64 | v x64]
// ---------------------------------------------------------------------------
__global__ __launch_bounds__(32)
void attn_kernel(const __bf16* __restrict__ qkv, __bf16* __restrict__ ao) {
  __shared__ __bf16 pbuf[16 * 32];   // P tile (16q x 32kv), A-frag staging
  __shared__ __bf16 vbuf[64 * 40];   // V^T tile [d][kv], stride 40 for 16B align

  int lane = threadIdx.x & 31;
  int ln = lane & 15, hh = lane >> 4;
  int q0 = blockIdx.x * 16;
  int head = blockIdx.y;
  int g = head >> 2, qi = head & 3;
  int qoff = g * 384 + qi * 64;
  int koff = g * 384 + 256;
  int voff = g * 384 + 320;
  const float scale = 0.125f;  // 1/sqrt(64)

  // Q fragments (K = 64 -> two 16x32 A-frags), loaded once
  v16bf qa[2];
  {
    const __bf16* p = qkv + (size_t)(q0 + ln) * QKVO + qoff;
    int kb = hh * 8;
#pragma unroll
    for (int kk = 0; kk < 2; kk++) {
      v8bf lo = *(const v8bf*)(p + kk * 32 + kb);
      v8bf hi = *(const v8bf*)(p + kk * 32 + kb + 16);
      qa[kk] = cat8(lo, hi);
    }
  }

  float mrow[8], lrow[8];
  v8f o[4] = {};
#pragma unroll
  for (int j = 0; j < 8; j++) { mrow[j] = -3.0e38f; lrow[j] = 0.f; }

  for (int t0 = 0; t0 <= q0; t0 += 32) {
    // ---- S = Q * K^T for 32 keys (two 16-wide N subtiles, K=64) ----
    v8f sfr[2] = {};
#pragma unroll
    for (int t = 0; t < 2; t++) {
      int kv = t0 + t * 16 + ln;
      if (kv > SEQ - 1) kv = SEQ - 1;
      const __bf16* kp = qkv + (size_t)kv * QKVO + koff + hh * 8;
#pragma unroll
      for (int kk = 0; kk < 2; kk++) {
        v16bf kb16 = cat8(*(const v8bf*)(kp + kk * 32), *(const v8bf*)(kp + kk * 32 + 16));
        sfr[t] = wmma_bf16(qa[kk], kb16, sfr[t]);
      }
    }
    // ---- scale + causal mask + online softmax ----
    float ps[2][8];
#pragma unroll
    for (int t = 0; t < 2; t++)
#pragma unroll
      for (int j = 0; j < 8; j++) {
        int r = q0 + j + 8 * hh;
        int c = t0 + t * 16 + ln;
        ps[t][j] = (c > r) ? -3.0e38f : sfr[t][j] * scale;
      }
#pragma unroll
    for (int j = 0; j < 8; j++) {
      float mt = fmaxf(ps[0][j], ps[1][j]);
#pragma unroll
      for (int off = 8; off >= 1; off >>= 1)
        mt = fmaxf(mt, __shfl_xor(mt, off, 32));   // stays within 16-lane half
      float mn = fmaxf(mrow[j], mt);
      float alpha = __expf(mrow[j] - mn);
      float rs = 0.f;
#pragma unroll
      for (int t = 0; t < 2; t++) { ps[t][j] = __expf(ps[t][j] - mn); rs += ps[t][j]; }
#pragma unroll
      for (int off = 8; off >= 1; off >>= 1)
        rs += __shfl_xor(rs, off, 32);
      mrow[j] = mn;
      lrow[j] = lrow[j] * alpha + rs;
#pragma unroll
      for (int t = 0; t < 4; t++) o[t][j] = o[t][j] * alpha;
    }
    // ---- stage P (C-frag -> A-frag layout) and V^T in LDS ----
#pragma unroll
    for (int t = 0; t < 2; t++)
#pragma unroll
      for (int j = 0; j < 8; j++)
        pbuf[(j + 8 * hh) * 32 + t * 16 + ln] = (__bf16)ps[t][j];
    {
      int vr = t0 + lane;
      if (vr > SEQ - 1) vr = SEQ - 1;
      const __bf16* vp = qkv + (size_t)vr * QKVO + voff;
#pragma unroll 8
      for (int n = 0; n < 64; n++) vbuf[n * 40 + lane] = vp[n];
    }
    __syncthreads();
    // ---- O += P(16x32) * V(32x64) ----
    v16bf pa;
    {
      const __bf16* pp = pbuf + ln * 32 + hh * 8;
      pa = cat8(*(const v8bf*)pp, *(const v8bf*)(pp + 16));
    }
#pragma unroll
    for (int t = 0; t < 4; t++) {
      const __bf16* vpp = vbuf + (size_t)(t * 16 + ln) * 40 + hh * 8;
      v16bf vb = cat8(*(const v8bf*)vpp, *(const v8bf*)(vpp + 16));
      o[t] = wmma_bf16(pa, vb, o[t]);
    }
    __syncthreads();
  }

#pragma unroll
  for (int t = 0; t < 4; t++)
#pragma unroll
    for (int j = 0; j < 8; j++) {
      float val = o[t][j] / lrow[j];
      ao[(size_t)(q0 + j + 8 * hh) * HID + head * 64 + t * 16 + ln] = (__bf16)val;
    }
}

// ---------------------------------------------------------------------------
// LayerNorm: one block per token, H=1024, 256 threads x 4 elements
// ---------------------------------------------------------------------------
__global__ __launch_bounds__(256)
void ln_kernel(const float* __restrict__ x, const float* __restrict__ w,
               const float* __restrict__ b, __bf16* __restrict__ obf,
               float* __restrict__ of) {
  __shared__ float r1[256], r2[256];
  __shared__ float s_mu, s_rstd;
  int row = blockIdx.x, tid = threadIdx.x;
  const float* xr = x + (size_t)row * HID;
  float v[4], s = 0.f, s2 = 0.f;
#pragma unroll
  for (int i = 0; i < 4; i++) {
    v[i] = xr[tid + i * 256];
    s += v[i]; s2 += v[i] * v[i];
  }
  r1[tid] = s; r2[tid] = s2;
  __syncthreads();
  for (int off = 128; off > 0; off >>= 1) {
    if (tid < off) { r1[tid] += r1[tid + off]; r2[tid] += r2[tid + off]; }
    __syncthreads();
  }
  if (tid == 0) {
    float mu = r1[0] * (1.0f / HID);
    float var = r2[0] * (1.0f / HID) - mu * mu;
    s_mu = mu; s_rstd = rsqrtf(var + 1e-5f);
  }
  __syncthreads();
  float mu = s_mu, rstd = s_rstd;
#pragma unroll
  for (int i = 0; i < 4; i++) {
    int h = tid + i * 256;
    float y = (v[i] - mu) * rstd * w[h] + b[h];
    obf[(size_t)row * HID + h] = (__bf16)y;
    if (of) of[(size_t)row * HID + h] = y;
  }
}

// ---------------------------------------------------------------------------
// Weight conversion kernels
// ---------------------------------------------------------------------------
__global__ void cvt_bf16_kernel(const float* __restrict__ in,
                                __bf16* __restrict__ out, size_t n) {
  size_t i = (size_t)blockIdx.x * blockDim.x + threadIdx.x;
  size_t stride = (size_t)gridDim.x * blockDim.x;
  for (; i < n; i += stride) out[i] = (__bf16)in[i];
}

// in: [mats][R][C] f32  ->  out: [mats][C][R] bf16
__global__ void cvt_transpose_kernel(const float* __restrict__ in,
                                     __bf16* __restrict__ out,
                                     int R, int C, int mats) {
  size_t total = (size_t)mats * R * C;
  size_t i = (size_t)blockIdx.x * blockDim.x + threadIdx.x;
  size_t stride = (size_t)gridDim.x * blockDim.x;
  size_t rc = (size_t)R * C;
  for (; i < total; i += stride) {
    size_t e = i / rc, rem = i % rc;
    size_t c = rem / R, r = rem % R;
    out[i] = (__bf16)in[e * rc + r * C + c];
  }
}

// ---------------------------------------------------------------------------
// Router: logits = ln2 @ rw (H x E), softmax, top-2, renormalize
// ---------------------------------------------------------------------------
__global__ void router_kernel(const float* __restrict__ x,
                              const float* __restrict__ rw,
                              int* __restrict__ topi, float* __restrict__ topp) {
  int tok = blockIdx.x * blockDim.x + threadIdx.x;
  if (tok >= SEQ) return;
  const float* xr = x + (size_t)tok * HID;
  float acc[NEXP] = {};
  for (int h = 0; h < HID; h++) {
    float xv = xr[h];
#pragma unroll
    for (int e = 0; e < NEXP; e++) acc[e] += xv * rw[h * NEXP + e];
  }
  float mx = acc[0];
#pragma unroll
  for (int e = 1; e < NEXP; e++) mx = fmaxf(mx, acc[e]);
  float p[NEXP], sum = 0.f;
#pragma unroll
  for (int e = 0; e < NEXP; e++) { p[e] = __expf(acc[e] - mx); sum += p[e]; }
#pragma unroll
  for (int e = 0; e < NEXP; e++) p[e] /= sum;
  int i0 = 0;
#pragma unroll
  for (int e = 1; e < NEXP; e++) if (p[e] > p[i0]) i0 = e;
  int i1 = (i0 == 0) ? 1 : 0;
#pragma unroll
  for (int e = 0; e < NEXP; e++) if (e != i0 && p[e] > p[i1]) i1 = e;
  float pn = p[i0] + p[i1];
  topi[tok * 2] = i0;      topi[tok * 2 + 1] = i1;
  topp[tok * 2] = p[i0] / pn; topp[tok * 2 + 1] = p[i1] / pn;
}

__global__ void init_moe_kernel(int* __restrict__ tok2slot,
                                int* __restrict__ slot_src,
                                float* __restrict__ slot_prob) {
  int i = blockIdx.x * blockDim.x + threadIdx.x;
  if (i < TSLOT) { slot_src[i] = -1; slot_prob[i] = 0.f; }
  if (i < SEQ * 2) tok2slot[i] = -1;
}

// Deterministic capacity assignment: one block per expert, token-order packing
// (matches the reference's stable sort-by-expert semantics).
__global__ __launch_bounds__(256)
void assign_kernel(const int* __restrict__ topi, const float* __restrict__ topp,
                   int* __restrict__ tok2slot, int* __restrict__ slot_src,
                   float* __restrict__ slot_prob) {
  __shared__ int scanbuf[256];
  __shared__ int basecnt;
  int e = blockIdx.x, tid = threadIdx.x;
  if (tid == 0) basecnt = 0;
  __syncthreads();
  for (int c0 = 0; c0 < SEQ; c0 += 256) {
    int tok = c0 + tid;
    int kidx = -1; float p = 0.f;
    if (topi[tok * 2] == e)          { kidx = 0; p = topp[tok * 2]; }
    else if (topi[tok * 2 + 1] == e) { kidx = 1; p = topp[tok * 2 + 1]; }
    int flag = (kidx >= 0) ? 1 : 0;
    scanbuf[tid] = flag;
    __syncthreads();
    for (int off = 1; off < 256; off <<= 1) {
      int v = (tid >= off) ? scanbuf[tid - off] : 0;
      __syncthreads();
      scanbuf[tid] += v;
      __syncthreads();
    }
    int pos = basecnt + scanbuf[tid] - 1;
    if (flag && pos < CAPE) {
      int slot = e * CAPE + pos;
      slot_src[slot] = tok;
      slot_prob[slot] = p;
      tok2slot[tok * 2 + kidx] = slot;
    }
    __syncthreads();
    if (tid == 255) basecnt += scanbuf[255];
    __syncthreads();
  }
}

__global__ __launch_bounds__(256)
void gather_kernel(const __bf16* __restrict__ ln2bf,
                   const int* __restrict__ slot_src,
                   __bf16* __restrict__ ptok) {
  int slot = blockIdx.x, tid = threadIdx.x;
  int src = slot_src[slot];
  __bf16* dst = ptok + (size_t)slot * HID;
  if (src >= 0) {
    const __bf16* s = ln2bf + (size_t)src * HID;
#pragma unroll
    for (int i = 0; i < 4; i++) dst[tid + i * 256] = s[tid + i * 256];
  } else {
#pragma unroll
    for (int i = 0; i < 4; i++) dst[tid + i * 256] = (__bf16)0.f;
  }
}

__global__ __launch_bounds__(256)
void combine_kernel(const float* __restrict__ h1, const float* __restrict__ fc2,
                    const int* __restrict__ tok2slot,
                    const float* __restrict__ slot_prob,
                    float* __restrict__ out) {
  int s = blockIdx.x, tid = threadIdx.x;
  int sl0 = tok2slot[s * 2], sl1 = tok2slot[s * 2 + 1];
  float p0 = (sl0 >= 0) ? slot_prob[sl0] : 0.f;
  float p1 = (sl1 >= 0) ? slot_prob[sl1] : 0.f;
#pragma unroll
  for (int i = 0; i < 4; i++) {
    int h = tid + i * 256;
    float v = h1[(size_t)s * HID + h];
    if (sl0 >= 0) v += p0 * fc2[(size_t)sl0 * HID + h];
    if (sl1 >= 0) v += p1 * fc2[(size_t)sl1 * HID + h];
    out[(size_t)s * HID + h] = v;
  }
}

// ---------------------------------------------------------------------------
// Host launcher
// ---------------------------------------------------------------------------
extern "C" void kernel_launch(void* const* d_in, const int* in_sizes, int n_in,
                              void* d_out, int out_size, void* d_ws, size_t ws_size,
                              hipStream_t stream) {
  const float* hidden  = (const float*)d_in[0];
  const float* ln1_w   = (const float*)d_in[1];
  const float* ln1_b   = (const float*)d_in[2];
  const float* ln2_w   = (const float*)d_in[3];
  const float* ln2_b   = (const float*)d_in[4];
  const float* qkv_w   = (const float*)d_in[5];
  const float* proj_w  = (const float*)d_in[6];
  const float* rout_w  = (const float*)d_in[7];
  const float* moe_w1  = (const float*)d_in[8];
  const float* moe_w2  = (const float*)d_in[9];
  float* out = (float*)d_out;

  // Workspace carve-out (256B aligned slices)
  char* base = (char*)d_ws;
  size_t off = 0;
  auto take = [&](size_t bytes) -> char* {
    char* p = base + off;
    off += (bytes + 255) & ~(size_t)255;
    return p;
  };
  __bf16* qkvw_bf = (__bf16*)take((size_t)QKVO * HID * 2);
  __bf16* projw_bf = (__bf16*)take((size_t)HID * HID * 2);
  __bf16* w1t = (__bf16*)take((size_t)NEXP * FFD * HID * 2);  // [E][F][H]
  __bf16* w2t = (__bf16*)take((size_t)NEXP * HID * FFD * 2);  // [E][H][F]
  __bf16* ln1_bf = (__bf16*)take((size_t)SEQ * HID * 2);
  __bf16* qkv_bf = (__bf16*)take((size_t)SEQ * QKVO * 2);
  __bf16* ao_bf  = (__bf16*)take((size_t)SEQ * HID * 2);
  float*  h1     = (float*)take((size_t)SEQ * HID * 4);
  __bf16* ln2_bf = (__bf16*)take((size_t)SEQ * HID * 2);
  float*  ln2_f  = (float*)take((size_t)SEQ * HID * 4);
  int*    topi   = (int*)take((size_t)SEQ * 2 * 4);
  float*  topp   = (float*)take((size_t)SEQ * 2 * 4);
  int*    tok2slot = (int*)take((size_t)SEQ * 2 * 4);
  int*    slot_src = (int*)take((size_t)TSLOT * 4);
  float*  slot_prob = (float*)take((size_t)TSLOT * 4);
  __bf16* ptok   = (__bf16*)take((size_t)TSLOT * HID * 2);
  __bf16* fc1_bf = (__bf16*)take((size_t)TSLOT * FFD * 2);
  float*  fc2_f  = (float*)take((size_t)TSLOT * HID * 4);
  (void)ws_size; (void)in_sizes; (void)n_in; (void)out_size;

  const int BIG = 1 << 30;

  // 1) Weight conversions (bf16 for the WMMA pipes)
  cvt_bf16_kernel<<<1024, 256, 0, stream>>>(qkv_w, qkvw_bf, (size_t)QKVO * HID);
  cvt_bf16_kernel<<<1024, 256, 0, stream>>>(proj_w, projw_bf, (size_t)HID * HID);
  cvt_transpose_kernel<<<2048, 256, 0, stream>>>(moe_w1, w1t, HID, FFD, NEXP);
  cvt_transpose_kernel<<<2048, 256, 0, stream>>>(moe_w2, w2t, FFD, HID, NEXP);
  init_moe_kernel<<<(TSLOT + 255) / 256, 256, 0, stream>>>(tok2slot, slot_src, slot_prob);

  // 2) LN1 -> QKV GEMM (bf16 out)
  ln_kernel<<<SEQ, 256, 0, stream>>>(hidden, ln1_w, ln1_b, ln1_bf, nullptr);
  gemm64x64<2><<<dim3(SEQ / 64, QKVO / 64), 128, 0, stream>>>(
      ln1_bf, qkvw_bf, HID, HID, qkv_bf, QKVO, HID, nullptr, BIG, 0);

  // 3) Causal GQA flash attention (WMMA)
  attn_kernel<<<dim3(SEQ / 16, NHEAD), 32, 0, stream>>>(qkv_bf, ao_bf);

  // 4) Output proj + residual: h1 = hidden + ao @ Wproj^T
  gemm64x64<1><<<dim3(SEQ / 64, HID / 64), 128, 0, stream>>>(
      ao_bf, projw_bf, HID, HID, h1, HID, HID, hidden, BIG, 0);

  // 5) LN2, router, capacity assignment, gather
  ln_kernel<<<SEQ, 256, 0, stream>>>(h1, ln2_w, ln2_b, ln2_bf, ln2_f);
  router_kernel<<<SEQ / 256, 256, 0, stream>>>(ln2_f, rout_w, topi, topp);
  assign_kernel<<<NEXP, 256, 0, stream>>>(topi, topp, tok2slot, slot_src, slot_prob);
  gather_kernel<<<TSLOT, 256, 0, stream>>>(ln2_bf, slot_src, ptok);

  // 6) MoE: fc1 (GELU, bf16 out) then fc2 (f32 out), per-expert B
  //    (CAPE=640 is a multiple of 64, so 64-row blocks never straddle experts)
  gemm64x64<3><<<dim3(TSLOT / 64, FFD / 64), 128, 0, stream>>>(
      ptok, w1t, HID, HID, fc1_bf, FFD, HID, nullptr, CAPE, (size_t)FFD * HID);
  gemm64x64<0><<<dim3(TSLOT / 64, HID / 64), 128, 0, stream>>>(
      fc1_bf, w2t, FFD, FFD, fc2_f, HID, FFD, nullptr, CAPE, (size_t)HID * FFD);

  // 7) Final combine: out = h1 + sum_k prob_k * fc2[slot_k]
  combine_kernel<<<SEQ, 256, 0, stream>>>(h1, fc2_f, tok2slot, slot_prob, out);
}
